// ModernNCA_32409823216325
// MI455X (gfx1250) — matrix operand order repaired and checked by reading, not compiled, not run on hardware
//
#include <hip/hip_runtime.h>

// ---------------------------------------------------------------------------
// ModernNCA forward for MI455X (gfx1250, wave32, WMMA bf16 16x16x32 + TDM)
// ---------------------------------------------------------------------------

typedef __bf16 bf16;
typedef __attribute__((ext_vector_type(8)))  __bf16 v8bf;
typedef __attribute__((ext_vector_type(16))) __bf16 v16bf;
typedef __attribute__((ext_vector_type(8)))  float  v8f;

#define B_ROWS   1024      // query rows
#define C_ROWS   100000    // candidates
#define C_PAD    100352    // 196 * 512
#define KDIM     512       // encoded dim
#define KENC     776       // 24*32 + 8
#define KP       800       // padded to multiple of 32
#define D_NUM    24
#define N_FREQ   16
#define D_OUT    10
#define TWO_PI   6.283185307179586f

// ---- workspace layout (bytes) ----
#define OFF_ENCWB  ((size_t)0)                         // 512*800 bf16
#define OFF_XE     (OFF_ENCWB + (size_t)512*KP*2)      // 1024*512 bf16
#define OFF_XNORM  (OFF_XE    + (size_t)B_ROWS*KDIM*2) // 1024 f32
#define OFF_CE     (OFF_XNORM + (size_t)B_ROWS*4)      // 100352*512 bf16
#define OFF_CNORM  (OFF_CE    + (size_t)C_PAD*KDIM*2)  // 100352 f32
#define OFF_GSUM   (OFF_CNORM + (size_t)C_PAD*4)       // 1024*10 f32

// ---------------------------------------------------------------------------
// Fragment loaders (this code's own contiguous-K order, same for A and B, not the ISA table order; wave32)
// A 16x32 bf16: lane = M%16 ; half-wave selects K-base 0/16 ; 16 contiguous K
// B 32x16 bf16: lane = N%16 ; half-wave selects K-base 0/16 ; 16 contiguous K
// ---------------------------------------------------------------------------
static __device__ inline v16bf ld_frag_a(const bf16* p) {
    v8bf lo = *(const v8bf*)(p);
    v8bf hi = *(const v8bf*)(p + 8);
    v16bf r;
#pragma unroll
    for (int i = 0; i < 8; ++i) { r[i] = lo[i]; r[i + 8] = hi[i]; }
    return r;
}
static __device__ inline v16bf ld_frag_b(const bf16* p) {
    v8bf lo = *(const v8bf*)(p);
    v8bf hi = *(const v8bf*)(p + 8);
    v16bf r;
#pragma unroll
    for (int i = 0; i < 8; ++i) { r[i] = lo[i]; r[i + 8] = hi[i]; }
    return r;
}
static __device__ inline v8f wmma_bf16(v16bf a, v16bf b, v8f c) {
    return __builtin_amdgcn_wmma_f32_16x16x32_bf16(
        /*neg_a=*/false, a, /*neg_b=*/false, b,
        /*c_mod=*/(short)0, c, /*reuse_a=*/false, /*reuse_b=*/false);
}
static __device__ inline void gatomic_addf(float* p, float v) {
#if defined(__gfx1250__)
    unsafeAtomicAdd(p, v);          // native global_atomic_add_f32
#else
    atomicAdd(p, v);
#endif
}

// ---------------------------------------------------------------------------
// enc_w (512 x 776 f32, row-major) -> bf16 (512 x 800), zero-padded K
// ---------------------------------------------------------------------------
__global__ __launch_bounds__(256) void k_convert_encw(const float* __restrict__ enc_w,
                                                      bf16* __restrict__ encwb) {
    int n = blockIdx.x;                       // 0..511
    for (int k = threadIdx.x; k < KP; k += 256)
        encwb[(size_t)n * KP + k] = (k < KENC) ? (bf16)enc_w[(size_t)n * KENC + k]
                                               : (bf16)0.0f;
}

// ---------------------------------------------------------------------------
// Encoder: 32 rows per block. Scalar PLR into LDS bf16 feature tile, then
// WMMA GEMM feat(32x800) x enc_w^T(800x512) + bias, bf16 store + row norms.
// ---------------------------------------------------------------------------
__global__ __launch_bounds__(256) void k_encode(const float* __restrict__ v,
                                                const float* __restrict__ freq,
                                                const float* __restrict__ plrw,
                                                const float* __restrict__ plrb,
                                                const bf16*  __restrict__ encwb,
                                                const float* __restrict__ encb,
                                                bf16*  __restrict__ out,
                                                float* __restrict__ norms) {
    __shared__ float vt[32 * 32];          // raw input tile
    __shared__ float freqL[D_NUM * N_FREQ];
    __shared__ float plrwL[32 * 32];
    __shared__ float plrbL[32];
    __shared__ bf16  featL[32 * KP];       // 32 x 800 bf16 (51200 B)
    __shared__ float lnorm[32];

    const int tid  = threadIdx.x;
    const int lane = tid & 31;
    const int wave = tid >> 5;             // 0..7
    const int l15  = lane & 15;
    const int hl   = lane >> 4;
    const int r0   = blockIdx.x * 32;

    // stage constants + input tile
    for (int i = tid; i < 32 * 32; i += 256) vt[i] = v[(size_t)r0 * 32 + i];
    for (int i = tid; i < D_NUM * N_FREQ; i += 256) freqL[i] = freq[i];
    for (int i = tid; i < 32 * 32; i += 256) plrwL[i] = plrw[i];
    if (tid < 32) { plrbL[tid] = plrb[tid]; lnorm[tid] = 0.0f; }
    __syncthreads();

    // --- PLR scalar encode: 768 (row,feature) items over 256 threads ---
    for (int item = tid; item < 32 * D_NUM; item += 256) {
        int r = item / D_NUM, f = item % D_NUM;
        float xv = vt[r * 32 + f];
        float cs[N_FREQ], sn[N_FREQ];
#pragma unroll
        for (int k = 0; k < N_FREQ; ++k) {
            float ang = TWO_PI * freqL[f * N_FREQ + k] * xv;
            __sincosf(ang, &sn[k], &cs[k]);
        }
        for (int j = 0; j < 32; ++j) {
            float a = plrbL[j];
#pragma unroll
            for (int k = 0; k < N_FREQ; ++k)
                a += plrwL[j * 32 + k] * cs[k] + plrwL[j * 32 + 16 + k] * sn[k];
            featL[r * KP + f * 32 + j] = (bf16)fmaxf(a, 0.0f);
        }
    }
    // categorical passthrough + zero pad
    for (int i = tid; i < 32 * 8; i += 256) {
        int r = i >> 3, c = i & 7;
        featL[r * KP + 768 + c] = (bf16)vt[r * 32 + D_NUM + c];
    }
    for (int i = tid; i < 32 * (KP - KENC); i += 256) {
        int r = i / (KP - KENC), p = i % (KP - KENC);
        featL[r * KP + KENC + p] = (bf16)0.0f;
    }
    __syncthreads();

    // --- WMMA GEMM: 2 M-subtiles x 4 N-subtiles per wave, 25 K-steps ---
    const int wbase = wave * 64;           // this wave's 64 output columns
    v8f acc[2][4];
#pragma unroll
    for (int mt = 0; mt < 2; ++mt)
#pragma unroll
        for (int nt = 0; nt < 4; ++nt) acc[mt][nt] = (v8f){0,0,0,0,0,0,0,0};

    for (int kk = 0; kk < KP / 32; ++kk) {
        const int kbA = kk * 32 + (hl << 4);
        v16bf a0 = ld_frag_a(&featL[(0  + l15) * KP + kbA]);
        v16bf a1 = ld_frag_a(&featL[(16 + l15) * KP + kbA]);
        const int kbB = kk * 32 + (hl << 4);
#pragma unroll
        for (int nt = 0; nt < 4; ++nt) {
            const bf16* bp = encwb + (size_t)(wbase + nt * 16 + l15) * KP + kbB;
            v16bf b = ld_frag_b(bp);
            acc[0][nt] = wmma_bf16(a0, b, acc[0][nt]);
            acc[1][nt] = wmma_bf16(a1, b, acc[1][nt]);
        }
    }

    // --- epilogue: bias, store bf16, accumulate row norms ---
    float rn[2][8];
#pragma unroll
    for (int mt = 0; mt < 2; ++mt)
#pragma unroll
        for (int j = 0; j < 8; ++j) rn[mt][j] = 0.0f;

#pragma unroll
    for (int mt = 0; mt < 2; ++mt) {
#pragma unroll
        for (int nt = 0; nt < 4; ++nt) {
            int col = wbase + nt * 16 + l15;
            float bias = encb[col];
#pragma unroll
            for (int j = 0; j < 8; ++j) {
                int row = mt * 16 + j + (hl << 3);
                float val = acc[mt][nt][j] + bias;
                out[(size_t)(r0 + row) * KDIM + col] = (bf16)val;
                rn[mt][j] += val * val;
            }
        }
    }
#pragma unroll
    for (int mt = 0; mt < 2; ++mt) {
#pragma unroll
        for (int j = 0; j < 8; ++j) {
            float s = rn[mt][j];
            s += __shfl_xor(s, 1, 16);
            s += __shfl_xor(s, 2, 16);
            s += __shfl_xor(s, 4, 16);
            s += __shfl_xor(s, 8, 16);
            if (l15 == 0) atomicAdd(&lnorm[mt * 16 + j + (hl << 3)], s);
        }
    }
    __syncthreads();
    if (tid < 32) norms[r0 + tid] = lnorm[tid];
}

// ---------------------------------------------------------------------------
// Distance GEMM + class-bucketed exp accumulation.
// Block = 32 query rows (blockIdx.y) x 512 candidates (blockIdx.x).
// xe tile staged into LDS via the Tensor Data Mover when available.
// ---------------------------------------------------------------------------
__global__ __launch_bounds__(256) void k_nca(const bf16*  __restrict__ xe,
                                             const float* __restrict__ xnorm,
                                             const bf16*  __restrict__ ce,
                                             const float* __restrict__ cnorm,
                                             const int*   __restrict__ candy,
                                             float* __restrict__ gsum) {
    __shared__ bf16  xeT[32 * KDIM];       // 32 KB
    __shared__ float xnL[32];
    __shared__ float ls[32 * D_OUT];       // class sums for this row tile

    const int tid  = threadIdx.x;
    const int lane = tid & 31;
    const int wave = tid >> 5;
    const int l15  = lane & 15;
    const int hl   = lane >> 4;
    const int rtile = blockIdx.y;          // 0..31
    const int candbase = blockIdx.x * 512 + wave * 64;

    // ---- stage xe tile: TDM 2D descriptor load (512x32 elem, 2B) ----
#if __has_builtin(__builtin_amdgcn_tensor_load_to_lds)
    if (wave == 0) {
        typedef __attribute__((ext_vector_type(4))) unsigned int u32x4;
        typedef __attribute__((ext_vector_type(8))) int i32x8;
        typedef __attribute__((ext_vector_type(4))) int i32x4;
        unsigned long long ga =
            (unsigned long long)(uintptr_t)(xe + (size_t)rtile * 32 * KDIM);
        unsigned int ldso = (unsigned int)(uintptr_t)&xeT[0];
        // D# group0: count=1 | lds_addr | global_addr[56:0] | type=2
        u32x4 g0 = { 1u, ldso,
                     (unsigned int)(ga & 0xffffffffu),
                     (unsigned int)((ga >> 32) & 0x01ffffffu) | (2u << 30) };
        // D# group1: data_size=2B; tensor 512x32; tile 512x32; stride0=512
        i32x8 g1 = { (int)(1u << 16),      // wg_mask=0, data_size=1 (2 bytes)
                     (int)(512u << 16),    // tensor_dim0[15:0]=512 @bits[63:48]
                     (int)(32u  << 16),    // tensor_dim0 hi=0 | tensor_dim1=32
                     (int)(512u << 16),    // tensor_dim1 hi=0 | tile_dim0=512
                     (int)32,              // tile_dim1=32 | tile_dim2=0
                     (int)512,             // tensor_dim0_stride[31:0]=512
                     0, 0 };               // stride0 hi | tensor_dim1_stride=0
        i32x4 gz = { 0, 0, 0, 0 };
#if __clang_major__ >= 23
        i32x8 gz8 = { 0, 0, 0, 0, 0, 0, 0, 0 };
        __builtin_amdgcn_tensor_load_to_lds(g0, g1, gz, gz, gz8, 0);
#else
        __builtin_amdgcn_tensor_load_to_lds(g0, g1, gz, gz, 0);
#endif
        __builtin_amdgcn_s_wait_tensorcnt(0);
    }
#else
    {
        const uint4* src = (const uint4*)(xe + (size_t)rtile * 32 * KDIM);
        uint4* dst = (uint4*)xeT;
        for (int i = tid; i < 32 * KDIM / 8; i += 256) dst[i] = src[i];
    }
#endif
    if (tid < 32) xnL[tid] = xnorm[rtile * 32 + tid];
    for (int i = tid; i < 32 * D_OUT; i += 256) ls[i] = 0.0f;

    // prefetch candidate chunk (512 cands x 1 KiB = 4096 lines)
    {
        const char* cb = (const char*)(ce + (size_t)blockIdx.x * 512 * KDIM);
        for (int i = tid; i < 4096; i += 256) __builtin_prefetch(cb + i * 128, 0, 1);
    }
    __syncthreads();

    // --- WMMA GEMM: 2 M-subtiles x 4 N-subtiles per wave, 16 K-steps ---
    v8f acc[2][4];
#pragma unroll
    for (int mt = 0; mt < 2; ++mt)
#pragma unroll
        for (int nt = 0; nt < 4; ++nt) acc[mt][nt] = (v8f){0,0,0,0,0,0,0,0};

    for (int kk = 0; kk < KDIM / 32; ++kk) {
        const int kbA = kk * 32 + (hl << 4);
        v16bf a0 = ld_frag_a(&xeT[(0  + l15) * KDIM + kbA]);
        v16bf a1 = ld_frag_a(&xeT[(16 + l15) * KDIM + kbA]);
        const int kbB = kk * 32 + (hl << 4);
#pragma unroll
        for (int nt = 0; nt < 4; ++nt) {
            const bf16* bp = ce + (size_t)(candbase + nt * 16 + l15) * KDIM + kbB;
            v16bf b = ld_frag_b(bp);
            acc[0][nt] = wmma_bf16(a0, b, acc[0][nt]);
            acc[1][nt] = wmma_bf16(a1, b, acc[1][nt]);
        }
    }

    // --- epilogue: dist -> exp(-dist) -> class buckets (LDS atomics) ---
#pragma unroll
    for (int nt = 0; nt < 4; ++nt) {
        int cand = candbase + nt * 16 + l15;
        float cn = cnorm[cand];
        int y = (cand < C_ROWS) ? candy[cand] : -1;
#pragma unroll
        for (int mt = 0; mt < 2; ++mt) {
#pragma unroll
            for (int j = 0; j < 8; ++j) {
                int row = mt * 16 + j + (hl << 3);
                float dot = acc[mt][nt][j];
                float sq  = xnL[row] + cn - 2.0f * dot;
                float d   = sqrtf(fmaxf(sq, 1e-12f));
                float e   = __expf(-d);        // TEMP = 1, max-shift = 0 (s <= 0)
                if (y >= 0) atomicAdd(&ls[row * D_OUT + y], e);
            }
        }
    }
    __syncthreads();

    // merge block partials into global class sums (native f32 atomics)
    for (int i = tid; i < 32 * D_OUT; i += 256)
        gatomic_addf(&gsum[(size_t)rtile * 32 * D_OUT + i], ls[i]);
}

// ---------------------------------------------------------------------------
// out[i][c] = log( gsum[i][c] / sum_c(gsum[i][*]) + EPS )
// ---------------------------------------------------------------------------
__global__ __launch_bounds__(256) void k_final(const float* __restrict__ gsum,
                                               float* __restrict__ out) {
    int i = blockIdx.x * 256 + threadIdx.x;
    if (i >= B_ROWS) return;
    float g[D_OUT], tot = 0.0f;
#pragma unroll
    for (int c = 0; c < D_OUT; ++c) { g[c] = gsum[i * D_OUT + c]; tot += g[c]; }
    float inv = (tot > 0.0f) ? (1.0f / tot) : 0.0f;
#pragma unroll
    for (int c = 0; c < D_OUT; ++c)
        out[i * D_OUT + c] = logf(g[c] * inv + 1e-7f);
}

// ---------------------------------------------------------------------------
extern "C" void kernel_launch(void* const* d_in, const int* in_sizes, int n_in,
                              void* d_out, int out_size, void* d_ws, size_t ws_size,
                              hipStream_t stream) {
    const float* x    = (const float*)d_in[0];   // 1024 x 32
    const float* cx   = (const float*)d_in[1];   // 100000 x 32
    const int*   cy   = (const int*)  d_in[2];   // 100000
    const float* freq = (const float*)d_in[3];   // 24 x 16
    const float* plrw = (const float*)d_in[4];   // 32 x 32
    const float* plrb = (const float*)d_in[5];   // 32
    const float* encw = (const float*)d_in[6];   // 512 x 776
    const float* encb = (const float*)d_in[7];   // 512
    (void)in_sizes; (void)n_in; (void)out_size; (void)ws_size;

    char* ws = (char*)d_ws;
    bf16*  encwb = (bf16*)(ws + OFF_ENCWB);
    bf16*  xe    = (bf16*)(ws + OFF_XE);
    float* xn    = (float*)(ws + OFF_XNORM);
    bf16*  ce    = (bf16*)(ws + OFF_CE);
    float* cn    = (float*)(ws + OFF_CNORM);
    float* gsum  = (float*)(ws + OFF_GSUM);

    // zero: global class sums + candidate pad rows/norms (deterministic per call)
    hipMemsetAsync(gsum, 0, (size_t)B_ROWS * D_OUT * 4, stream);
    hipMemsetAsync((char*)ce + (size_t)C_ROWS * KDIM * 2, 0,
                   (size_t)(C_PAD - C_ROWS) * KDIM * 2, stream);
    hipMemsetAsync(cn + C_ROWS, 0, (size_t)(C_PAD - C_ROWS) * 4, stream);

    k_convert_encw<<<512, 256, 0, stream>>>(encw, encwb);
    k_encode<<<B_ROWS / 32, 256, 0, stream>>>(x,  freq, plrw, plrb, encwb, encb, xe, xn);
    k_encode<<<C_ROWS / 32, 256, 0, stream>>>(cx, freq, plrw, plrb, encwb, encb, ce, cn);
    k_nca<<<dim3(C_PAD / 512, B_ROWS / 32), 256, 0, stream>>>(xe, xn, ce, cn, cy, gsum);
    k_final<<<(B_ROWS + 255) / 256, 256, 0, stream>>>(gsum, (float*)d_out);
}
